// MultiChainCondenseMSA_g_4329327034444
// MI455X (gfx1250) — matrix operand order repaired and checked by reading, not compiled, not run on hardware
//
#include <hip/hip_runtime.h>

// CDNA5 / gfx1250: wave32, v_wmma_f32_16x16x32_bf16 (f32 accumulate).
typedef __bf16 v16bf __attribute__((ext_vector_type(16)));
typedef float  v8f   __attribute__((ext_vector_type(8)));

union Frag {
  v16bf v;
  unsigned short u[16];
  uint4 q[2];                 // two 16-byte halves -> ds_load_b128 / global_load_b128
};

__device__ __forceinline__ unsigned short f2bf(float f) {
  unsigned int u = __float_as_uint(f);
  u += 0x7FFFu + ((u >> 16) & 1u);          // round-to-nearest-even (cold paths only)
  return (unsigned short)(u >> 16);
}

// Pack two f32 -> two bf16 in one u32. Prefer the HW packed converter.
__device__ __forceinline__ unsigned int pack_bf16x2(float lo, float hi) {
#if __has_builtin(__builtin_amdgcn_cvt_pk_bf16_f32)
  auto p = __builtin_amdgcn_cvt_pk_bf16_f32(lo, hi);   // v_cvt_pk_bf16_f32
  unsigned int r;
  __builtin_memcpy(&r, &p, 4);
  return r;
#else
  unsigned int a = __float_as_uint(lo) + 0x8000u;      // round-half-up, 1 VALU each
  unsigned int b = __float_as_uint(hi) + 0x8000u;
  return (a >> 16) | (b & 0xFFFF0000u);
#endif
}

constexpr int Bc = 8, Tc = 30, Lc = 12, Ac = 50, Fc = 32, Hc = 128;
constexpr int LF = Lc * Fc;   // 384
constexpr int KP = 64;        // padded K (A-dim) for GEMM1
constexpr int CS = LF;        // Csh row stride (768 B, 16B-aligned rows)

// Yt element offset: [row][a] with 8-element XOR bank swizzle (16B granules).
__device__ __forceinline__ int yoff(int row, int a) {
  return row * KP + ((((a >> 3) ^ (row & 7)) << 3) | (a & 7));
}

// W [1024,128] f32 -> Wt [128][1024] bf16 (transposed: K contiguous per h).
__global__ void wcvt_kernel(const float* __restrict__ W,
                            unsigned short* __restrict__ Wt) {
  int idx = blockIdx.x * 256 + threadIdx.x;     // idx = h*1024 + k
  if (idx < Fc * Fc * Hc) {
    int h = idx >> 10, k = idx & 1023;
    Wt[idx] = f2bf(W[k * Hc + h]);
  }
}

__global__ __launch_bounds__(256, 1)
void condense_cov_kernel(const float* __restrict__ matches,
                         const float* __restrict__ rmsds,
                         const float* __restrict__ mask,
                         const unsigned short* __restrict__ Wt,
                         float* __restrict__ out) {
  __shared__ __align__(16) unsigned short Ysh[LF * KP];  // wc^T bf16 [l*F+f][a]  48 KB
  __shared__ __align__(16) unsigned short Csh[16 * CS];  // half cov strip        12 KB
  __shared__ __align__(16) unsigned short zb[32];        // 64B zero block (j>=12 lanes)
  __shared__ float wts[Lc * Ac];
  __shared__ float mean[Lc * Fc];
  __shared__ float mloc[Lc];

  const int bt    = blockIdx.x;              // one workgroup per (b,t)
  const int b     = bt / Tc;
  const int t     = bt - b * Tc;
  const int tid   = threadIdx.x;
  const int lane  = tid & 31;
  const int wave  = tid >> 5;                // 8 waves
  const int hal   = lane >> 4;
  const int lanlo = lane & 15;

  const long baseR = (long)(b * (Tc * Lc) + t * Lc) * Ac;
  const long baseM = baseR * Fc;

  if (tid == 0) __builtin_prefetch(matches + baseM, 0, 1);   // global_prefetch_b8
  if (tid < 32) zb[tid] = 0;

  // ---- softmax(-rmsd) over A per residue (threads 0..11) + mask row ----
  if (tid < Lc) {
    const int l = tid;
    const float* rp = rmsds + baseR + (long)l * Ac;
    float mx = -3.4e38f;
    for (int a = 0; a < Ac; ++a) mx = fmaxf(mx, -rp[a]);
    float s = 0.f;
    for (int a = 0; a < Ac; ++a) { float e = __expf(-rp[a] - mx); wts[l * Ac + a] = e; s += e; }
    float inv = 1.f / s;
    for (int a = 0; a < Ac; ++a) wts[l * Ac + a] *= inv;
    mloc[l] = mask[(b * Tc + t) * Lc + l];
  }
  __syncthreads();

  // ---- weighted mean [L,F] ----
  for (int idx = tid; idx < Lc * Fc; idx += 256) {
    int l = idx >> 5;
    const float* mp = matches + baseM + (long)l * Ac * Fc + (idx & 31);
    float s = 0.f;
    for (int a = 0; a < Ac; ++a) s += wts[l * Ac + a] * mp[(long)a * Fc];
    mean[idx] = s;
  }
  __syncthreads();

  // ---- Yt[l*F+f][a] = bf16(w*(x-mean)), a in [0,64) zero-padded, packed b32 stores
  for (int idx = tid; idx < LF * (KP / 2); idx += 256) {
    int pr = idx / LF;                 // a-pair index (consecutive tid -> consecutive lf)
    int lf = idx - pr * LF;
    int a0 = pr << 1;
    int l = lf >> 5, f = lf & 31;
    float v0 = 0.f, v1 = 0.f;
    if (a0 < Ac)
      v0 = wts[l * Ac + a0]     * (matches[baseM + ((long)l * Ac + a0) * Fc + f]     - mean[lf]);
    if (a0 + 1 < Ac)
      v1 = wts[l * Ac + a0 + 1] * (matches[baseM + ((long)l * Ac + a0 + 1) * Fc + f] - mean[lf]);
    *(unsigned int*)&Ysh[yoff(lf, a0)] = pack_bf16x2(v0, v1);
  }
  __syncthreads();

  for (int i = 0; i < Lc; ++i) {
    v8f acc2 = {};                     // GEMM2 accumulator; N-tile == wave id
    const int h = 16 * wave + lanlo;   // hidden index (H=128 = 8 waves * 16)

    for (int phase = 0; phase < 2; ++phase) {
      // ===== GEMM1 half: cov rows f = 16*phase..+15 : Yi^T(32x64) x Y(64x384) =====
      const int rowA = i * Fc + 16 * phase + lanlo;   // A-frag M row (f)
      for (int q3 = 0; q3 < 3; ++q3) {                // 24 N-tiles over 8 waves
        const int colB = 16 * (wave * 3 + q3) + lanlo;
        v8f acc1 = {};
        #pragma unroll
        for (int ks = 0; ks < 2; ++ks) {
          Frag Af, Bf;
          Af.q[0] = *(const uint4*)&Ysh[yoff(rowA, 32 * ks + 16 * hal)];      // K runs per
          Af.q[1] = *(const uint4*)&Ysh[yoff(rowA, 32 * ks + 16 * hal + 8)];  // A layout
          Bf.q[0] = *(const uint4*)&Ysh[yoff(colB, 32 * ks + 16 * hal)];      // sequential
          Bf.q[1] = *(const uint4*)&Ysh[yoff(colB, 32 * ks + 16 * hal + 8)];  // B layout
          acc1 = __builtin_amdgcn_wmma_f32_16x16x32_bf16(false, Af.v, false, Bf.v,
                                                         (short)0, acc1, false, false);
        }
        const float me = mloc[i] * mloc[colB >> 5];   // mask_edges(i,j)
        #pragma unroll
        for (int r = 0; r < 8; r += 2) {              // packed f32->bf16 conversion
          unsigned int pk = pack_bf16x2(acc1[r] * me, acc1[r + 1] * me);
          Csh[(r     + 8 * hal) * CS + colB] = (unsigned short)pk;
          Csh[(r + 1 + 8 * hal) * CS + colB] = (unsigned short)(pk >> 16);
        }
      }
      __syncthreads();

      // ===== GEMM2 half: [16(j),1024] x Wt, K-chunk ks reads one Csh row =====
      const int j = lanlo;
      for (int ksl = 0; ksl < 16; ++ksl) {
        const int ks = 16 * phase + ksl;
        Frag Af, Bf;
        const unsigned short* pj = (j < Lc) ? &Csh[ksl * CS + j * Fc] : zb;
        Af.q[0] = *(const uint4*)(pj + 16 * hal);
        Af.q[1] = *(const uint4*)(pj + 16 * hal + 8);
        const unsigned short* pw = Wt + ((long)h << 10) + 32 * ks + 16 * hal;
        Bf.q[0] = *(const uint4*)pw;
        Bf.q[1] = *(const uint4*)(pw + 8);
        acc2 = __builtin_amdgcn_wmma_f32_16x16x32_bf16(false, Af.v, false, Bf.v,
                                                       (short)0, acc2, false, false);
      }
      __syncthreads();
    }

    // ---- store out[b,t,i,j,h] (f32), j<12 valid ----
    const long obase = (((long)bt * Lc + i) * Lc) * Hc;
    #pragma unroll
    for (int r = 0; r < 8; ++r) {
      int jj = r + 8 * hal;
      if (jj < Lc) out[obase + (long)jj * Hc + h] = acc2[r];
    }
  }
}

extern "C" void kernel_launch(void* const* d_in, const int* in_sizes, int n_in,
                              void* d_out, int out_size, void* d_ws, size_t ws_size,
                              hipStream_t stream) {
  const float* matches = (const float*)d_in[0];
  const float* rmsds   = (const float*)d_in[1];
  const float* mask    = (const float*)d_in[2];
  const float* W       = (const float*)d_in[3];
  // d_in[4] = term_len (== 12, compiled in)

  unsigned short* Wt = (unsigned short*)d_ws;      // 1024*128*2 = 256 KB scratch
  const int nW = Fc * Fc * Hc;                     // 131072
  wcvt_kernel<<<(nW + 255) / 256, 256, 0, stream>>>(W, Wt);
  condense_cov_kernel<<<Bc * Tc, 256, 0, stream>>>(matches, rmsds, mask, Wt,
                                                   (float*)d_out);
}
